// NNUE_78683800862979
// MI455X (gfx1250) — compile-verified, matrix operand order checked by reading, not run on hardware
//
#include <hip/hip_runtime.h>

typedef __attribute__((ext_vector_type(16))) _Float16 v16h;
typedef __attribute__((ext_vector_type(8)))  _Float16 v8h;
typedef __attribute__((ext_vector_type(8)))  float    v8f;

#define IN_F   40960
#define BATCH  4096
#define HID    256
#define TILE_M 128
#define TILE_N 128
#define TILE_K 32
#define LDSS   40   // halves per LDS row: 32 data + 8 pad (80 B = 5x16 B)

union V16U { v16h v; v8h h[2]; };

// ---------------------------------------------------------------------------
// Kernel 1: fused feature-transformer GEMM, software-pipelined.
//   C[8192 x 256] = concat(white, black)[8192 x 40960] * W_ft^T[40960 x 256]
//   clip to [0,1], scatter into x[4096 x 512]  (cols 0..255 white, 256..511 black)
// Pipeline: global loads (k+1) -> regs | WMMA on LDS[cur] | cvt+store regs ->
// LDS[nxt] | barrier.  One barrier per k-step, double-buffered LDS.
// Global addresses held as per-thread pointers bumped by TILE_K each step.
// ---------------------------------------------------------------------------
__global__ __launch_bounds__(256)
void nnue_ft_gemm(const float* __restrict__ white,
                  const float* __restrict__ black,
                  const float* __restrict__ Wft,
                  float* __restrict__ xout)
{
    __shared__ alignas(16) _Float16 ldsA[2][TILE_M * LDSS];
    __shared__ alignas(16) _Float16 ldsB[2][TILE_N * LDSS];

    const int tid  = threadIdx.x;
    const int wave = tid >> 5;
    const int lane = tid & 31;

    const int  mBlock  = blockIdx.x;          // 0..63 : 32 white blocks, 32 black
    const bool isBlack = mBlock >= 32;
    const float* Abase = isBlack ? black : white;
    const int mBase = (mBlock & 31) * TILE_M;
    const int nBase = blockIdx.y * TILE_N;

    const int waveM = (wave & 3) * 32;        // 4 waves down M
    const int waveN = (wave >> 2) * 64;       // 2 waves across N

    // Staging map: float4 id f = tid + i*256 -> row f>>3, col4 (f&7)*4.
    // Persistent per-thread global pointers, bumped by TILE_K each k-step.
    const float* aPtr[4];
    const float* bPtr[4];
    int sr[4], sk[4];
#pragma unroll
    for (int i = 0; i < 4; ++i) {
        const int f = tid + i * 256;
        sr[i] = f >> 3;
        sk[i] = (f & 7) << 2;
        aPtr[i] = Abase + (size_t)(mBase + sr[i]) * IN_F + sk[i];
        bPtr[i] = Wft   + (size_t)(nBase + sr[i]) * IN_F + sk[i];
    }

    v8f acc[2][4];
#pragma unroll
    for (int mi = 0; mi < 2; ++mi)
#pragma unroll
        for (int ni = 0; ni < 4; ++ni)
            acc[mi][ni] = (v8f){0.f,0.f,0.f,0.f,0.f,0.f,0.f,0.f};

    // WMMA 16-bit operand layout (ISA 7.12.2): lane L -> row/col (L&15),
    // halves 0..7 = K kb..kb+7, halves 8..15 = K 16+kb..16+kb+7, kb = L<16?0:8
    const int rsel  = lane & 15;
    const int kbase = (lane < 16) ? 0 : 8;

    auto computeTile = [&](int buf) {
        V16U af[2], bf[4];
#pragma unroll
        for (int mi = 0; mi < 2; ++mi) {
            const int r = waveM + mi * 16 + rsel;
            af[mi].h[0] = *(const v8h*)&ldsA[buf][r * LDSS + kbase];
            af[mi].h[1] = *(const v8h*)&ldsA[buf][r * LDSS + 16 + kbase];
        }
#pragma unroll
        for (int ni = 0; ni < 4; ++ni) {
            const int c = waveN + ni * 16 + rsel;
            bf[ni].h[0] = *(const v8h*)&ldsB[buf][c * LDSS + kbase];
            bf[ni].h[1] = *(const v8h*)&ldsB[buf][c * LDSS + 16 + kbase];
        }
#pragma unroll
        for (int mi = 0; mi < 2; ++mi)
#pragma unroll
            for (int ni = 0; ni < 4; ++ni)
                acc[mi][ni] = __builtin_amdgcn_wmma_f32_16x16x32_f16(
                    false, af[mi].v, false, bf[ni].v,
                    (short)0, acc[mi][ni], false, false);
    };

    float4 pa[4], pb[4];

    // ---- prologue: stage tile 0 into LDS[0] ----
#pragma unroll
    for (int i = 0; i < 4; ++i) {
        pa[i] = *(const float4*)aPtr[i];  aPtr[i] += TILE_K;
        pb[i] = *(const float4*)bPtr[i];  bPtr[i] += TILE_K;
    }
#pragma unroll
    for (int i = 0; i < 4; ++i) {
        _Float16* da = &ldsA[0][sr[i] * LDSS + sk[i]];
        da[0] = (_Float16)pa[i].x; da[1] = (_Float16)pa[i].y;
        da[2] = (_Float16)pa[i].z; da[3] = (_Float16)pa[i].w;
        _Float16* db = &ldsB[0][sr[i] * LDSS + sk[i]];
        db[0] = (_Float16)pb[i].x; db[1] = (_Float16)pb[i].y;
        db[2] = (_Float16)pb[i].z; db[3] = (_Float16)pb[i].w;
    }
    __syncthreads();

    int cur = 0;
    for (int k0 = TILE_K; k0 < IN_F; k0 += TILE_K) {
        const int nxt = cur ^ 1;

        // issue global loads for tile k0 (complete while WMMAs run),
        // plus a speculative L2 prefetch one tile further for the A stream
#pragma unroll
        for (int i = 0; i < 4; ++i) {
            pa[i] = *(const float4*)aPtr[i];
            __builtin_prefetch(aPtr[i] + TILE_K, 0, 3);
            aPtr[i] += TILE_K;
            pb[i] = *(const float4*)bPtr[i];
            bPtr[i] += TILE_K;
        }

        // compute on the already-staged buffer
        computeTile(cur);

        // convert + store the prefetched tile into the other buffer
#pragma unroll
        for (int i = 0; i < 4; ++i) {
            _Float16* da = &ldsA[nxt][sr[i] * LDSS + sk[i]];
            da[0] = (_Float16)pa[i].x; da[1] = (_Float16)pa[i].y;
            da[2] = (_Float16)pa[i].z; da[3] = (_Float16)pa[i].w;
            _Float16* db = &ldsB[nxt][sr[i] * LDSS + sk[i]];
            db[0] = (_Float16)pb[i].x; db[1] = (_Float16)pb[i].y;
            db[2] = (_Float16)pb[i].z; db[3] = (_Float16)pb[i].w;
        }
        __syncthreads();
        cur = nxt;
    }
    computeTile(cur);  // last staged tile

    // ---- epilogue: clip(0,1) and scatter into x[4096 x 512] ----
    // C/D layout: VGPR v, lane L -> M = v + (L<16?0:8), N = L&15
    const int rowAdd = (lane < 16) ? 0 : 8;
    const int colOff = nBase + waveN + rsel + (isBlack ? HID : 0);
#pragma unroll
    for (int mi = 0; mi < 2; ++mi) {
#pragma unroll
        for (int ni = 0; ni < 4; ++ni) {
#pragma unroll
            for (int v = 0; v < 8; ++v) {
                float d = acc[mi][ni][v];
                d = fminf(fmaxf(d, 0.0f), 1.0f);
                const int row = mBase + waveM + mi * 16 + v + rowAdd;
                const int col = colOff + ni * 16;
                xout[(size_t)row * (2 * HID) + col] = d;
            }
        }
    }
}

// ---------------------------------------------------------------------------
// Kernel 2: tiny MLP head. One wave32 per batch row.
// ---------------------------------------------------------------------------
__global__ __launch_bounds__(256)
void nnue_mlp(const float* __restrict__ x,
              const float* __restrict__ stm,
              const float* __restrict__ W1, const float* __restrict__ b1,
              const float* __restrict__ W2, const float* __restrict__ b2,
              const float* __restrict__ W3, const float* __restrict__ b3,
              float* __restrict__ out)
{
    const int wave = threadIdx.x >> 5;
    const int lane = threadIdx.x & 31;
    const int row  = blockIdx.x * 8 + wave;
    if (row >= BATCH) return;

    const float* xr  = x  + (size_t)row  * 512;
    const float* w1r = W1 + (size_t)lane * 512;

    float acc = b1[lane];
#pragma unroll 4
    for (int k = 0; k < 512; k += 4) {
        const float4 xv = *(const float4*)(xr + k);
        const float4 wv = *(const float4*)(w1r + k);
        acc = fmaf(xv.x, wv.x, acc);
        acc = fmaf(xv.y, wv.y, acc);
        acc = fmaf(xv.z, wv.z, acc);
        acc = fmaf(xv.w, wv.w, acc);
    }
    const float h1 = fmaxf(acc, 0.0f);

    float acc2 = b2[lane];
    const float* w2r = W2 + lane * 32;
#pragma unroll
    for (int j = 0; j < 32; ++j) {
        const float hj = __shfl(h1, j, 32);
        acc2 = fmaf(hj, w2r[j], acc2);
    }
    const float h2 = fmaxf(acc2, 0.0f);

    float p = h2 * W3[lane];
#pragma unroll
    for (int off = 16; off > 0; off >>= 1)
        p += __shfl_xor(p, off, 32);

    if (lane == 0)
        out[row] = (p + b3[0]) * stm[row];
}

// ---------------------------------------------------------------------------
extern "C" void kernel_launch(void* const* d_in, const int* in_sizes, int n_in,
                              void* d_out, int out_size, void* d_ws, size_t ws_size,
                              hipStream_t stream) {
    const float* white = (const float*)d_in[0];
    const float* black = (const float*)d_in[1];
    const float* stm   = (const float*)d_in[2];
    const float* Wft   = (const float*)d_in[3];
    const float* W1    = (const float*)d_in[4];
    const float* b1    = (const float*)d_in[5];
    const float* W2    = (const float*)d_in[6];
    const float* b2    = (const float*)d_in[7];
    const float* W3    = (const float*)d_in[8];
    const float* b3    = (const float*)d_in[9];

    float* x = (float*)d_ws;   // [4096 x 512] activations, 8 MB

    dim3 grid1(64, 2);         // 64 M-blocks (white+black) x 2 N-blocks
    nnue_ft_gemm<<<grid1, 256, 0, stream>>>(white, black, Wft, x);

    nnue_mlp<<<512, 256, 0, stream>>>(x, stm, W1, b1, W2, b2, W3, b3,
                                      (float*)d_out);
}
